// MoE_13572096655871
// MI455X (gfx1250) — matrix operand order repaired and checked by reading, not compiled, not run on hardware
//
#include <hip/hip_runtime.h>
#include <hip/hip_bf16.h>
#include <math.h>

// ---------------- problem constants ----------------
#define NTOK   8192      // B*S
#define DDIM   1024
#define FDIM   4096
#define NEXP   8
#define TOPK   2
#define CAP    1280      // int(8192/8 * 1.25)

// ---------------- vector types ----------------
typedef __attribute__((ext_vector_type(16))) __bf16 bf16x16;
typedef __attribute__((ext_vector_type(8)))  __bf16 bf16x8;
typedef __attribute__((ext_vector_type(4)))  __bf16 bf16x4;
typedef __attribute__((ext_vector_type(8)))  float  f32x8;
typedef __attribute__((ext_vector_type(4)))  float  f32x4;
typedef __attribute__((ext_vector_type(4)))  int    v4i;

typedef __attribute__((address_space(1))) v4i* gv4i_p;  // global <4 x i32>*
typedef __attribute__((address_space(3))) v4i* lv4i_p;  // LDS    <4 x i32>*

union AFrag { bf16x16 v; bf16x8 h[2]; };

__device__ __forceinline__ f32x8 wmma_bf16(bf16x16 a, bf16x16 b, f32x8 c) {
  return __builtin_amdgcn_wmma_f32_16x16x32_bf16(
      /*neg_a=*/false, a, /*neg_b=*/false, b,
      /*c_mod=*/(short)0, c, /*reuse_a=*/false, /*reuse_b=*/false);
}

// ---- CDNA5 async global->LDS 16B copy (ASYNCcnt-tracked) ----
__device__ __forceinline__ void async_copy_b128(void* lds_ptr, const void* gptr) {
#if __has_builtin(__builtin_amdgcn_global_load_async_to_lds_b128)
  __builtin_amdgcn_global_load_async_to_lds_b128(
      (gv4i_p)gptr, (lv4i_p)lds_ptr, 0, 0);
#else
  // generic->LDS flat pointer: low 32 bits are the LDS offset (ISA §10.2)
  unsigned lds_off = (unsigned)(uintptr_t)lds_ptr;
  unsigned long long ga = (unsigned long long)(uintptr_t)gptr;
  asm volatile("global_load_async_to_lds_b128 %0, %1, off"
               :: "v"(lds_off), "v"(ga) : "memory");
#endif
}

__device__ __forceinline__ void wait_async0() {
#if __has_builtin(__builtin_amdgcn_s_wait_asynccnt)
  __builtin_amdgcn_s_wait_asynccnt(0);
#else
  asm volatile("s_wait_asynccnt 0x0" ::: "memory");
#endif
}

// =====================================================================
// 1) Router: logits -> softmax -> top2 (renormed) + aux-loss statistics
//    one wave per token (wave32)
// =====================================================================
__global__ __launch_bounds__(256) void router_kernel(
    const float* __restrict__ x, const float* __restrict__ gw,
    int* __restrict__ tidx, float* __restrict__ tw, float* __restrict__ stats)
{
  const int lane  = threadIdx.x & 31;
  const int wv    = threadIdx.x >> 5;
  const int token = blockIdx.x * 8 + wv;
  if (token >= NTOK) return;

  float acc[NEXP];
#pragma unroll
  for (int e = 0; e < NEXP; e++) acc[e] = 0.f;

  const float* xr = x + (size_t)token * DDIM;
  for (int d = lane; d < DDIM; d += 32) {
    float xv = xr[d];
#pragma unroll
    for (int e = 0; e < NEXP; e++) acc[e] += xv * gw[e * DDIM + d];
  }
#pragma unroll
  for (int e = 0; e < NEXP; e++) {
#pragma unroll
    for (int off = 16; off > 0; off >>= 1) acc[e] += __shfl_xor(acc[e], off);
  }

  if (lane == 0) {
    float m = acc[0];
#pragma unroll
    for (int e = 1; e < NEXP; e++) m = fmaxf(m, acc[e]);
    float p[NEXP], s = 0.f;
#pragma unroll
    for (int e = 0; e < NEXP; e++) { p[e] = __expf(acc[e] - m); s += p[e]; }
    float inv = __builtin_amdgcn_rcpf(s);
#pragma unroll
    for (int e = 0; e < NEXP; e++) p[e] *= inv;
    float lse = m + __logf(s);

    // top-2, ties -> lowest index (strict >)
    float b1 = -1.f, b2 = -1.f; int i1 = 0, i2 = 0;
#pragma unroll
    for (int e = 0; e < NEXP; e++) {
      if (p[e] > b1)      { b2 = b1; i2 = i1; b1 = p[e]; i1 = e; }
      else if (p[e] > b2) { b2 = p[e]; i2 = e; }
    }
    float rden = __builtin_amdgcn_rcpf(b1 + b2);
    tidx[token * 2 + 0] = i1;  tidx[token * 2 + 1] = i2;
    tw[token * 2 + 0] = b1 * rden;  tw[token * 2 + 1] = b2 * rden;

#pragma unroll
    for (int e = 0; e < NEXP; e++) atomicAdd(&stats[e], p[e]);   // mean_probs sums
    atomicAdd(&stats[NEXP + i1], 1.f);                           // density counts
    atomicAdd(&stats[NEXP + i2], 1.f);
    atomicAdd(&stats[2 * NEXP], lse * lse);                      // z-loss sum
  }
}

__global__ void finalize_kernel(const float* __restrict__ stats, float* __restrict__ out)
{
  const float Nf = (float)NTOK;
  float lb = 0.f;
  for (int e = 0; e < NEXP; e++)
    lb += (stats[NEXP + e] / (Nf * (float)TOPK)) * (stats[e] / Nf);
  lb *= 0.01f * (float)NEXP;                  // LOAD_BALANCE_WEIGHT * E
  float z = 0.001f * (stats[2 * NEXP] / Nf);  // ROUTER_Z_WEIGHT * mean(lse^2)
  out[(size_t)NTOK * DDIM] = lb + z;          // aux appended after flattened out
}

// =====================================================================
// 2) Order-exact capacity positions: one wave per (expert, k)
// =====================================================================
__global__ void pos_scan_kernel(const int* __restrict__ tidx, int* __restrict__ pos)
{
  const int e = blockIdx.x, k = blockIdx.y, lane = threadIdx.x;
  int base = 0;
  for (int t0 = 0; t0 < NTOK; t0 += 32) {
    int t = t0 + lane;
    bool active = (tidx[t * 2 + k] == e);
    unsigned mm = (unsigned)__ballot(active);
    int prefix = __popc(mm & ((1u << lane) - 1u));
    if (active) pos[t * 2 + k] = base + prefix;
    base += __popc(mm);
  }
}

// =====================================================================
// 3) Gather tokens into dispatch buffer (fp32 -> bf16)
// =====================================================================
__global__ __launch_bounds__(256) void gather_kernel(
    const float* __restrict__ x, const int* __restrict__ tidx,
    const int* __restrict__ pos, __bf16* __restrict__ disp, int k)
{
  const int t = blockIdx.x;
  const int e = tidx[t * 2 + k];
  const int p = pos[t * 2 + k];
  if (p >= CAP) return;  // dropped token (block-uniform)
  const float* src = x + (size_t)t * DDIM;
  __bf16* dst = disp + ((size_t)e * CAP + p) * DDIM;
  const int d0 = threadIdx.x * 4;
  f32x4 v = *(const f32x4*)(src + d0);
  bf16x4 b;
#pragma unroll
  for (int j = 0; j < 4; j++) b[j] = (__bf16)v[j];
  *(bf16x4*)(dst + d0) = b;
}

// =====================================================================
// 4) GEMM1 fused GeGLU: h = silu(disp @ wi_gate^T) * (disp @ wi_up^T)
//    block tile 128x(64x2), K-step 32, double-buffered LDS,
//    A staged via global_load_async_to_lds_b128, B cvt fp32->bf16 in VGPRs
// =====================================================================
__global__ __launch_bounds__(256) void gemm1_geglu_kernel(
    const __bf16* __restrict__ disp, const float* __restrict__ wgate,
    const float* __restrict__ wup, __bf16* __restrict__ hbuf)
{
  constexpr int BM = 128, BN = 64, BK = 32, KP = 40;  // KP: conflict-free pad
  __shared__ __bf16 As[2][BM * KP];
  __shared__ __bf16 Bs[2][2][BN * KP];

  const int e  = blockIdx.z;
  const int m0 = blockIdx.x * BM;
  const int n0 = blockIdx.y * BN;
  const int tid  = threadIdx.x;
  const int lane = tid & 31, wv = tid >> 5;
  const int half = lane >> 4, mr = lane & 15;

  const __bf16* A  = disp  + (size_t)e * CAP  * DDIM;
  const float*  Wg = wgate + (size_t)e * FDIM * DDIM;
  const float*  Wu = wup   + (size_t)e * FDIM * DDIM;

  // staging coordinates
  const int ar = tid >> 1, ac = (tid & 1) * 16;        // A: 2 thr/row, 16 halves
  const int bm = tid >> 7, bt = tid & 127;             // B: gate|up half
  const int br = bt >> 1,  bc = (bt & 1) * 16;         // 2 thr/row, 16 floats
  const float* Wsel = (bm == 0) ? Wg : Wu;

  auto stage = [&](int kb, int buf) {
    // A tile: async byte copy (bf16 already)
    const __bf16* srcA = A + (size_t)(m0 + ar) * DDIM + kb + ac;
    __bf16* dstA = &As[buf][ar * KP + ac];
    async_copy_b128(dstA, srcA);
    async_copy_b128(dstA + 8, srcA + 8);
    // B tile: fp32 load + cvt -> bf16 -> LDS
    const float* srcB = Wsel + (size_t)(n0 + br) * DDIM + kb + bc;
    f32x4 f0 = *(const f32x4*)(srcB);
    f32x4 f1 = *(const f32x4*)(srcB + 4);
    f32x4 f2 = *(const f32x4*)(srcB + 8);
    f32x4 f3 = *(const f32x4*)(srcB + 12);
    bf16x8 lo, hi;
#pragma unroll
    for (int j = 0; j < 4; j++) { lo[j] = (__bf16)f0[j]; lo[4 + j] = (__bf16)f1[j]; }
#pragma unroll
    for (int j = 0; j < 4; j++) { hi[j] = (__bf16)f2[j]; hi[4 + j] = (__bf16)f3[j]; }
    *(bf16x8*)&Bs[buf][bm][br * KP + bc]     = lo;
    *(bf16x8*)&Bs[buf][bm][br * KP + bc + 8] = hi;
  };

  f32x8 accG[4], accU[4];
#pragma unroll
  for (int i = 0; i < 4; i++)
#pragma unroll
    for (int j = 0; j < 8; j++) { accG[i][j] = 0.f; accU[i][j] = 0.f; }

  stage(0, 0);
  wait_async0();
  __syncthreads();

  for (int kb = 0; kb < DDIM; kb += BK) {
    const int cur = (kb >> 5) & 1;
    if (kb + BK < DDIM) stage(kb + BK, cur ^ 1);  // prefetch next tile

    AFrag a;  // A: row per lane, contiguous K16 per half (matches B)
    a.h[0] = *(const bf16x8*)&As[cur][(wv * 16 + mr) * KP + half * 16];
    a.h[1] = *(const bf16x8*)&As[cur][(wv * 16 + mr) * KP + half * 16 + 8];
#pragma unroll
    for (int nt = 0; nt < 4; nt++) {
      const int bn = nt * 16 + mr;
      AFrag b;  // B: column per lane, contiguous K16 per half
      b.h[0] = *(const bf16x8*)&Bs[cur][0][bn * KP + half * 16];
      b.h[1] = *(const bf16x8*)&Bs[cur][0][bn * KP + half * 16 + 8];
      accG[nt] = wmma_bf16(a.v, b.v, accG[nt]);
      b.h[0] = *(const bf16x8*)&Bs[cur][1][bn * KP + half * 16];
      b.h[1] = *(const bf16x8*)&Bs[cur][1][bn * KP + half * 16 + 8];
      accU[nt] = wmma_bf16(a.v, b.v, accU[nt]);
    }
    wait_async0();      // next tile's async copies landed
    __syncthreads();    // everyone done reading cur / writing cur^1
  }

  __bf16* H = hbuf + (size_t)e * CAP * FDIM;
#pragma unroll
  for (int nt = 0; nt < 4; nt++) {
    const int col = n0 + nt * 16 + mr;
#pragma unroll
    for (int r = 0; r < 8; r++) {           // C/D layout: M = r + half*8
      const int row = m0 + wv * 16 + r + half * 8;
      float g = accG[nt][r], u = accU[nt][r];
      float hv = g * __builtin_amdgcn_rcpf(1.f + __expf(-g)) * u;   // silu(g)*u
      H[(size_t)row * FDIM + col] = (__bf16)hv;
    }
  }
}

// =====================================================================
// 5) GEMM2: eout = h @ wo^T   A:(CAP x FDIM) bf16, B:(DDIM x FDIM) fp32
//    block tile 128x128, double-buffered, async A staging
// =====================================================================
__global__ __launch_bounds__(256) void gemm2_kernel(
    const __bf16* __restrict__ hbuf, const float* __restrict__ wo,
    float* __restrict__ eout)
{
  constexpr int BM = 128, BN = 128, BK = 32, KP = 40;
  __shared__ __bf16 As[2][BM * KP];
  __shared__ __bf16 Bs[2][BN * KP];

  const int e  = blockIdx.z;
  const int m0 = blockIdx.x * BM;
  const int n0 = blockIdx.y * BN;
  const int tid  = threadIdx.x;
  const int lane = tid & 31, wv = tid >> 5;
  const int half = lane >> 4, mr = lane & 15;

  const __bf16* A = hbuf + (size_t)e * CAP  * FDIM;
  const float*  W = wo   + (size_t)e * DDIM * FDIM;

  const int ar = tid >> 1, ac = (tid & 1) * 16;   // A: 2 thr/row
  const int br = tid >> 1, bc = (tid & 1) * 16;   // B: 128 rows, 2 thr/row

  auto stage = [&](int kb, int buf) {
    const __bf16* srcA = A + (size_t)(m0 + ar) * FDIM + kb + ac;
    __bf16* dstA = &As[buf][ar * KP + ac];
    async_copy_b128(dstA, srcA);
    async_copy_b128(dstA + 8, srcA + 8);
    const float* srcB = W + (size_t)(n0 + br) * FDIM + kb + bc;
    f32x4 f0 = *(const f32x4*)(srcB);
    f32x4 f1 = *(const f32x4*)(srcB + 4);
    f32x4 f2 = *(const f32x4*)(srcB + 8);
    f32x4 f3 = *(const f32x4*)(srcB + 12);
    bf16x8 lo, hi;
#pragma unroll
    for (int j = 0; j < 4; j++) { lo[j] = (__bf16)f0[j]; lo[4 + j] = (__bf16)f1[j]; }
#pragma unroll
    for (int j = 0; j < 4; j++) { hi[j] = (__bf16)f2[j]; hi[4 + j] = (__bf16)f3[j]; }
    *(bf16x8*)&Bs[buf][br * KP + bc]     = lo;
    *(bf16x8*)&Bs[buf][br * KP + bc + 8] = hi;
  };

  f32x8 acc[8];
#pragma unroll
  for (int i = 0; i < 8; i++)
#pragma unroll
    for (int j = 0; j < 8; j++) acc[i][j] = 0.f;

  stage(0, 0);
  wait_async0();
  __syncthreads();

  for (int kb = 0; kb < FDIM; kb += BK) {
    const int cur = (kb >> 5) & 1;
    if (kb + BK < FDIM) stage(kb + BK, cur ^ 1);

    AFrag a;
    a.h[0] = *(const bf16x8*)&As[cur][(wv * 16 + mr) * KP + half * 16];
    a.h[1] = *(const bf16x8*)&As[cur][(wv * 16 + mr) * KP + half * 16 + 8];
#pragma unroll
    for (int nt = 0; nt < 8; nt++) {
      const int bn = nt * 16 + mr;
      AFrag b;
      b.h[0] = *(const bf16x8*)&Bs[cur][bn * KP + half * 16];
      b.h[1] = *(const bf16x8*)&Bs[cur][bn * KP + half * 16 + 8];
      acc[nt] = wmma_bf16(a.v, b.v, acc[nt]);
    }
    wait_async0();
    __syncthreads();
  }

  float* O = eout + (size_t)e * CAP * DDIM;
#pragma unroll
  for (int nt = 0; nt < 8; nt++) {
    const int col = n0 + nt * 16 + mr;
#pragma unroll
    for (int r = 0; r < 8; r++) {
      const int row = m0 + wv * 16 + r + half * 8;
      O[(size_t)row * DDIM + col] = acc[nt][r];
    }
  }
}

// =====================================================================
// 6) Combine: out[t] += eout[e, pos] * (weight * valid)
// =====================================================================
__global__ __launch_bounds__(256) void combine_kernel(
    const float* __restrict__ eout, const int* __restrict__ tidx,
    const int* __restrict__ pos, const float* __restrict__ tw,
    float* __restrict__ out, int k)
{
  const int t = blockIdx.x;
  const int e = tidx[t * 2 + k];
  const int p = pos[t * 2 + k];
  if (p >= CAP) return;                      // invalid -> contributes 0
  const float w = tw[t * 2 + k];
  const float* src = eout + ((size_t)e * CAP + p) * DDIM;
  float* dst = out + (size_t)t * DDIM;
  const int d0 = threadIdx.x * 4;
  f32x4 s = *(const f32x4*)(src + d0);
  f32x4 d = *(const f32x4*)(dst + d0);
  d += s * w;
  *(f32x4*)(dst + d0) = d;
}

// =====================================================================
// launcher
// =====================================================================
extern "C" void kernel_launch(void* const* d_in, const int* in_sizes, int n_in,
                              void* d_out, int out_size, void* d_ws, size_t ws_size,
                              hipStream_t stream) {
  const float* x       = (const float*)d_in[0];
  const float* gate_w  = (const float*)d_in[1];
  const float* wi_gate = (const float*)d_in[2];
  const float* wi_up   = (const float*)d_in[3];
  const float* wo      = (const float*)d_in[4];
  float* out = (float*)d_out;

  // workspace layout
  char* ws = (char*)d_ws;
  size_t off = 0;
  const size_t DISP_BYTES = (size_t)NEXP * CAP * DDIM * sizeof(__bf16);
  const size_t H_BYTES    = (size_t)NEXP * CAP * FDIM * sizeof(__bf16);
  const size_t EOUT_BYTES = (size_t)NEXP * CAP * DDIM * sizeof(float);
  __bf16* disp = (__bf16*)(ws + off); off += DISP_BYTES;
  __bf16* hbuf = (__bf16*)(ws + off); off += H_BYTES;
  float*  eout = (float*) (ws + off); off += EOUT_BYTES;
  int*    tidx = (int*)   (ws + off); off += (size_t)NTOK * 2 * sizeof(int);
  float*  tw   = (float*) (ws + off); off += (size_t)NTOK * 2 * sizeof(float);
  int*    pos  = (int*)   (ws + off); off += (size_t)NTOK * 2 * sizeof(int);
  float*  stats= (float*) (ws + off); off += 32 * sizeof(float);

  (void)hipMemsetAsync(stats, 0, 32 * sizeof(float), stream);
  (void)hipMemsetAsync(out, 0, (size_t)NTOK * DDIM * sizeof(float), stream);

  router_kernel<<<NTOK / 8, 256, 0, stream>>>(x, gate_w, tidx, tw, stats);
  finalize_kernel<<<1, 1, 0, stream>>>(stats, out);
  pos_scan_kernel<<<dim3(NEXP, TOPK), 32, 0, stream>>>(tidx, pos);

  for (int k = 0; k < TOPK; k++) {
    (void)hipMemsetAsync(disp, 0, DISP_BYTES, stream);
    gather_kernel<<<NTOK, 256, 0, stream>>>(x, tidx, pos, disp, k);
    gemm1_geglu_kernel<<<dim3(CAP / 128, FDIM / 64, NEXP), 256, 0, stream>>>(
        disp, wi_gate, wi_up, hbuf);
    gemm2_kernel<<<dim3(CAP / 128, DDIM / 128, NEXP), 256, 0, stream>>>(
        hbuf, wo, eout);
    combine_kernel<<<NTOK, 256, 0, stream>>>(eout, tidx, pos, tw, out, k);
  }
}
